// CNN1D_LSTM1_16853451670051
// MI455X (gfx1250) — matrix operand order repaired and checked by reading, not compiled, not run on hardware
//
#include <hip/hip_runtime.h>

// ---------- WMMA types / helpers (CDNA5 gfx1250, wave32) ----------
typedef __attribute__((ext_vector_type(16))) __bf16 v16bf;
typedef __attribute__((ext_vector_type(8)))  float  v8f;
typedef __attribute__((ext_vector_type(4)))  unsigned int u32x4;
typedef __attribute__((ext_vector_type(4)))  int v4i;

union BFrag {
  v16bf v;
  u32x4 q[2];
  unsigned int u[8];
  unsigned short s[16];
};

__device__ __forceinline__ unsigned short f2bf(float f) {
  union { float f; unsigned int u; } c; c.f = f;
  unsigned int u = c.u + 0x7FFFu + ((c.u >> 16) & 1u);   // RNE
  return (unsigned short)(u >> 16);
}
__device__ __forceinline__ float bf2f(unsigned short s) {
  union { unsigned int u; float f; } c; c.u = ((unsigned int)s) << 16;
  return c.f;
}
__device__ __forceinline__ v8f wmma_bf16(v16bf a, v16bf b, v8f c) {
  return __builtin_amdgcn_wmma_f32_16x16x32_bf16(false, a, false, b, (short)0, c, false, false);
}
__device__ __forceinline__ float lrelu(float x) { return x > 0.0f ? x : 0.01f * x; }
__device__ __forceinline__ float sigmf(float x) { return 1.0f / (1.0f + __expf(-x)); }

// ---------- CDNA5 async global->LDS DMA (ASYNCcnt), with safe fallback ----------
#if __has_builtin(__builtin_amdgcn_global_load_async_to_lds_b128)
#define HAVE_ASYNC_LDS 1
typedef __attribute__((address_space(1))) v4i gv4i;   // global-AS int4 ("__device__")
typedef __attribute__((address_space(3))) v4i sv4i;   // LDS-AS int4 ("__shared__")
__device__ __forceinline__ void async_copy_b128(void* lds, const void* g) {
  __builtin_amdgcn_global_load_async_to_lds_b128(
      (gv4i*)(void*)g, (sv4i*)lds, 0, 0);
}
__device__ __forceinline__ void async_wait0() {
#if __has_builtin(__builtin_amdgcn_s_wait_asynccnt)
  __builtin_amdgcn_s_wait_asynccnt(0);
#else
  asm volatile("s_wait_asynccnt 0x0" ::: "memory");
#endif
}
#else
#define HAVE_ASYNC_LDS 0
__device__ __forceinline__ void async_copy_b128(void* lds, const void* g) {
  *(u32x4*)lds = *(const u32x4*)g;   // synchronous fallback
}
__device__ __forceinline__ void async_wait0() {}
#endif

// ---------- problem constants ----------
#define BATCH 64
#define LIN   4096
#define L1    4067         // conv1 (k=30) output length
#define LP    811          // maxpool(k20,s5,ceil) output length
#define LPPAD 928          // padded bf16 pooled row stride (16B-aligned, covers tile overrun)
#define L2L   802          // conv2 (k=10) output length
#define BIN0  300
#define BIN1  100
// padded-K packings (taps rounded up so lanehi becomes a uniform address shift)
#define KC1   16           // conv1: 16 ch x 32 taps (30 real + 2 zero) = 512 = 16*32
#define KC2   16           // conv2: 32 ch x 16 taps (10 real + 6 zero) = 512 = 16*32

// ---------- workspace layout (bytes, all 256-aligned) ----------
static constexpr size_t O_BEFF = 0;                                   // 32 f32
static constexpr size_t O_W1B  = 256;                                 // [16][2][32][16] bf16
static constexpr size_t O_W2B  = O_W1B + (size_t)KC1*2*32*16*2;       // [16][4][32][16] bf16
static constexpr size_t O_WHH  = O_W2B + (size_t)KC2*4*32*16*2;       // [2][2][16][32][16] bf16
static constexpr size_t O_XBF  = O_WHH + (size_t)2*2*16*32*16*2;      // X as bf16 + pad
static constexpr size_t SZ_XBF = (size_t)BATCH*16*LIN*2 + 256;
static constexpr size_t O_PBF  = O_XBF + SZ_XBF;                      // pooled bf16, stride LPPAD
static constexpr size_t SZ_PBF = (size_t)BATCH*32*LPPAD*2;
static constexpr size_t O_FEAT = O_PBF + SZ_PBF;                      // 64*64*802 f32
static constexpr size_t SZ_FEAT = (size_t)BATCH*64*L2L*4;
static constexpr size_t O_XP0  = O_FEAT + SZ_FEAT;
static constexpr size_t O_XP1  = O_XP0 + (size_t)BATCH*64*BIN0*4;
static constexpr size_t O_SEQ0 = O_XP1 + (size_t)BATCH*64*BIN1*4;     // (300,64,4) f32
static constexpr size_t O_SEQ1 = O_SEQ0 + (size_t)BIN0*BATCH*4*4;     // (100,64,4) f32
static constexpr size_t O_H0   = O_SEQ1 + (size_t)BIN1*BATCH*4*4;     // 64*64 f32
static constexpr size_t O_H1   = O_H0 + (size_t)BATCH*64*4;

// =====================================================================
// K_xbf: one-time X fp32 -> bf16 (packed pairs)
// =====================================================================
__global__ __launch_bounds__(256) void k_xbf(const float* __restrict__ X,
                                             unsigned int* __restrict__ Xbf) {
  int idx = blockIdx.x * 256 + threadIdx.x;           // each handles 2 elements
  if (idx >= BATCH*16*LIN/2) return;
  float f0 = X[2*idx], f1 = X[2*idx + 1];
  Xbf[idx] = (unsigned int)f2bf(f0) | ((unsigned int)f2bf(f1) << 16);
}

// =====================================================================
// K_prep: effective bias + all WMMA B-matrix fragment packs (bf16).
// B-fragment layout: [kc][ntile][lane][e], kB = lanehi*8 + (e>>3)*16 + (e&7) (k within chunk),
//                    n = ntile*16 + (lane&15).
// conv1: kidx = ci*32 + t (t<30 real);  conv2: kidx = ci*16 + t (t<10 real).
// =====================================================================
__global__ __launch_bounds__(256) void k_prep(
    const float* __restrict__ w_dw, const float* __restrict__ b_dw,
    const float* __restrict__ w_pw, const float* __restrict__ b_pw,
    const float* __restrict__ w_c2,
    const float* __restrict__ w_hh0, const float* __restrict__ w_hh1,
    float* __restrict__ beff, unsigned short* __restrict__ w1B,
    unsigned short* __restrict__ w2B, unsigned short* __restrict__ whhB) {
  int idx = blockIdx.x * 256 + threadIdx.x;
  if (idx < 32) {
    float v = b_pw[idx];                           // dw bias folded through pointwise
    for (int c = 0; c < 256; ++c) v += w_pw[idx*256 + c] * b_dw[c];
    beff[idx] = v;
    return;
  }
  idx -= 32;
  if (idx < KC1*2*32*16) {  // fused conv1 weights W_eff, ci = kc, t = kB (pad t>=30 -> 0)
    int e = idx & 15, lane = (idx >> 4) & 31, nt = (idx >> 9) & 1, kc = idx >> 10;
    int kB = ((lane >> 4) << 3) + ((e >> 3) << 4) + (e & 7);
    int ci = kc, t = kB;
    int n = nt*16 + (lane & 15);
    float v = 0.0f;
    if (t < 30)
      for (int j = 0; j < 16; ++j)
        v += w_pw[n*256 + ci*16 + j] * w_dw[(ci*16 + j)*30 + t];
    w1B[idx] = f2bf(v);
    return;
  }
  idx -= KC1*2*32*16;
  if (idx < KC2*4*32*16) {  // conv2 weights, ci = 2kc + (kB>>4), t = kB&15 (pad t>=10 -> 0)
    int e = idx & 15, lane = (idx >> 4) & 31, nt = (idx >> 9) & 3, kc = idx >> 11;
    int kB = ((lane >> 4) << 3) + ((e >> 3) << 4) + (e & 7);
    int ci = kc*2 + (kB >> 4), t = kB & 15;
    int n = nt*16 + (lane & 15);
    w2B[idx] = (t < 10) ? f2bf(w_c2[(n*32 + ci)*10 + t]) : (unsigned short)0;
    return;
  }
  idx -= KC2*4*32*16;
  if (idx < 2*2*16*32*16) { // LSTM w_hh^T: B[k=hidden][g=gate], both branches
    int e = idx & 15, lane = (idx >> 4) & 31, nt = (idx >> 9) & 15;
    int kc = (idx >> 13) & 1, br = idx >> 14;
    int k = kc*32 + ((lane >> 4) << 3) + ((e >> 3) << 4) + (e & 7);
    int g = nt*16 + (lane & 15);
    const float* whh = br ? w_hh1 : w_hh0;
    whhB[idx] = f2bf(whh[g*64 + k]);
  }
}

// =====================================================================
// K_conv1: fused (dw+pw) conv 16->32 (32 padded taps) via WMMA + lrelu + maxpool
// Workgroup = (48 pool outputs, batch n) -> 256 conv positions. 8 waves x 2 M-tiles.
// A-element address = per-lane base + compile-time immediate (zero addr VALU).
// =====================================================================
__global__ __launch_bounds__(256) void k_conv1(
    const unsigned short* __restrict__ Xbf, const unsigned short* __restrict__ w1B,
    const float* __restrict__ beff, unsigned short* __restrict__ pooledbf) {
  __shared__ __align__(16) unsigned short Xs[16][288];  // bf16 input tile (halo 31)
  __shared__ float convS[32][256];                      // conv output (post-relu)
  __shared__ float beffs[32];

  const int tile = blockIdx.x, n = blockIdx.y;
  const int p0 = tile * 48;
  const int c0 = p0 * 5;                                // 240*tile -> 480B: 16B aligned
  const int tid = threadIdx.x;

  if (tid < 32) beffs[tid] = beff[tid];
  const unsigned short* Xn = Xbf + (size_t)n * 16 * LIN;
  // 16 rows x 36 groups of 8 bf16 (16B) each
  for (int i = tid; i < 16*36; i += 256) {
    int ci = i / 36, g = i - ci*36;
    async_copy_b128(&Xs[ci][g*8], Xn + ci*LIN + c0 + g*8);
  }
  if (tile + 1 < 17) __builtin_prefetch(Xn + c0 + 240, 0, 1);  // next tile
  async_wait0();
  __syncthreads();

  const int wave = tid >> 5, lane = tid & 31;
  const int lanehi = lane >> 4, lanelo = lane & 15;
  const u32x4* w1q = (const u32x4*)w1B;

  for (int cidx = 0; cidx < 2; ++cidx) {
    const int pml = (wave*2 + cidx) * 16;   // local position base of this M-tile
    const int prow = pml + lanelo;          // this lane's A row (position)
    const unsigned short* Xbase = &Xs[0][0] + prow + lanehi*8;  // per-lane base
    v8f acc0, acc1;
    #pragma unroll
    for (int r = 0; r < 8; ++r) { acc0[r] = beffs[lanelo]; acc1[r] = beffs[16 + lanelo]; }
    #pragma unroll
    for (int kc = 0; kc < KC1; ++kc) {      // K = 512 = 16 * 32 (fully unrolled)
      BFrag a;
      #pragma unroll
      for (int e = 0; e < 16; ++e)          // immediate-offset ds_load_u16
        a.s[e] = Xbase[kc*288 + (e >> 3)*16 + (e & 7)];
      BFrag b0, b1;
      int base0 = ((kc*2 + 0)*32 + lane)*2;
      int base1 = ((kc*2 + 1)*32 + lane)*2;
      b0.q[0] = w1q[base0]; b0.q[1] = w1q[base0 + 1];
      b1.q[0] = w1q[base1]; b1.q[1] = w1q[base1 + 1];
      acc0 = wmma_bf16(a.v, b0.v, acc0);
      acc1 = wmma_bf16(a.v, b1.v, acc1);
    }
    #pragma unroll
    for (int r = 0; r < 8; ++r) {           // D: m = r + 8*lanehi, n = lanelo
      int m = pml + r + 8*lanehi;
      bool oob = (c0 + m) >= L1;            // beyond conv length => -inf pad for pool
      convS[lanelo][m]      = oob ? -__builtin_inff() : lrelu(acc0[r]);
      convS[16 + lanelo][m] = oob ? -__builtin_inff() : lrelu(acc1[r]);
    }
  }
  __syncthreads();

  for (int i = tid; i < 48*32; i += 256) {
    int pp = i >> 5, ch = i & 31;
    int p = p0 + pp;
    if (p < LP) {
      float m = -__builtin_inff();
      #pragma unroll 4
      for (int j = 0; j < 20; ++j) m = fmaxf(m, convS[ch][pp*5 + j]);
      pooledbf[((size_t)n*32 + ch)*LPPAD + p] = f2bf(m);
    }
  }
}

// =====================================================================
// K_conv2: conv 32->64 (16 padded taps) via WMMA + lrelu. Workgroup = (128 pos, n).
// Pooled tile staged (bf16) via async LDS DMA; garbage beyond LP only feeds
// zero-weight taps / discarded D-rows.
// =====================================================================
__global__ __launch_bounds__(256) void k_conv2(
    const unsigned short* __restrict__ pooledbf, const unsigned short* __restrict__ w2B,
    const float* __restrict__ b_c2, float* __restrict__ feat) {
  __shared__ __align__(16) unsigned short Ps[32][144];
  __shared__ float bs[64];
  const int tileq = blockIdx.x, n = blockIdx.y;
  const int q0 = tileq * 128;                            // 256B-aligned offset
  const int tid = threadIdx.x;
  if (tid < 64) bs[tid] = b_c2[tid];
  const unsigned short* Pn = pooledbf + (size_t)n * 32 * LPPAD;
  // 32 rows x 18 groups of 8 bf16 (16B)
  for (int i = tid; i < 32*18; i += 256) {
    int c = i / 18, g = i - c*18;
    async_copy_b128(&Ps[c][g*8], Pn + c*LPPAD + q0 + g*8);
  }
  if (tileq + 1 < 7) __builtin_prefetch(Pn + q0 + 128, 0, 1);
  async_wait0();
  __syncthreads();

  const int wave = tid >> 5, lane = tid & 31;
  const int lanehi = lane >> 4, lanelo = lane & 15;
  const int pml = wave * 16;
  const int prow = pml + lanelo;
  const unsigned short* Pbase = &Ps[0][0] + prow + lanehi*8;   // per-lane base
  const u32x4* w2q = (const u32x4*)w2B;

  v8f acc[4];
  #pragma unroll
  for (int nt = 0; nt < 4; ++nt)
    #pragma unroll
    for (int r = 0; r < 8; ++r) acc[nt][r] = bs[nt*16 + lanelo];

  #pragma unroll
  for (int kc = 0; kc < KC2; ++kc) {        // K = 512 = 16 * 32 (fully unrolled)
    BFrag a;
    #pragma unroll
    for (int e = 0; e < 16; ++e)            // immediate-offset ds_load_u16
      a.s[e] = Pbase[kc*288 + (e >> 3)*144 + (e & 7)];
    #pragma unroll
    for (int nt = 0; nt < 4; ++nt) {
      BFrag b;
      int base = ((kc*4 + nt)*32 + lane)*2;
      b.q[0] = w2q[base]; b.q[1] = w2q[base + 1];
      acc[nt] = wmma_bf16(a.v, b.v, acc[nt]);
    }
  }
  #pragma unroll
  for (int nt = 0; nt < 4; ++nt) {
    int ch = nt*16 + lanelo;
    #pragma unroll
    for (int r = 0; r < 8; ++r) {
      int q = q0 + pml + r + 8*lanehi;
      if (q < L2L) feat[((size_t)n*64 + ch)*L2L + q] = lrelu(acc[nt][r]);
    }
  }
}

// =====================================================================
// K_apool: adaptive maxpool for both branches (b0: s=2,k=204; b1: s=8,k=10)
// =====================================================================
__global__ __launch_bounds__(256) void k_apool(
    const float* __restrict__ feat, float* __restrict__ xp0, float* __restrict__ xp1) {
  int idx = blockIdx.x * 256 + threadIdx.x;
  const int N0 = BATCH*64*BIN0;
  const int NT = N0 + BATCH*64*BIN1;
  if (idx >= NT) return;
  int s, k, bin; float* xp;
  if (idx < N0) { bin = BIN0; s = 2; k = 204; xp = xp0; }
  else          { idx -= N0; bin = BIN1; s = 8; k = 10;  xp = xp1; }
  int u = idx % bin; int nc = idx / bin; int c = nc % 64; int n = nc / 64;
  const float* f = feat + ((size_t)n*64 + c)*L2L + u*s;
  float m = -__builtin_inff();
  for (int j = 0; j < k; ++j) m = fmaxf(m, f[j]);
  xp[((size_t)n*64 + c)*bin + u] = m;
}

// =====================================================================
// K_conv3: branch conv 64->4 k=3 pad=1 + leaky_relu, output (T, B, 4)
// =====================================================================
__global__ __launch_bounds__(256) void k_conv3(
    const float* __restrict__ xp0, const float* __restrict__ xp1,
    const float* __restrict__ w_sc0, const float* __restrict__ b_sc0,
    const float* __restrict__ w_sc1, const float* __restrict__ b_sc1,
    float* __restrict__ seq0, float* __restrict__ seq1) {
  int idx = blockIdx.x * 256 + threadIdx.x;
  const int N0 = BATCH*4*BIN0;
  const int NT = N0 + BATCH*4*BIN1;
  if (idx >= NT) return;
  int bin; const float *xp, *w, *b; float* seq;
  if (idx < N0) { bin = BIN0; xp = xp0; w = w_sc0; b = b_sc0; seq = seq0; }
  else          { idx -= N0; bin = BIN1; xp = xp1; w = w_sc1; b = b_sc1; seq = seq1; }
  int u = idx % bin; int nc = idx / bin; int cc = nc % 4; int n = nc / 4;
  float acc = b[cc];
  for (int c = 0; c < 64; ++c) {
    const float* xr = xp + ((size_t)n*64 + c)*bin;
    const float* wr = w + (cc*64 + c)*3;
    #pragma unroll
    for (int dt = 0; dt < 3; ++dt) {
      int pos = u - 1 + dt;
      if (pos >= 0 && pos < bin) acc += xr[pos] * wr[dt];
    }
  }
  seq[((size_t)u*BATCH + n)*4 + cc] = lrelu(acc);
}

// =====================================================================
// K_lstm: one workgroup per branch. Per step: gates(64x256) = xg + h @ w_hh^T
// via WMMA (M=64 batch, K=64 hidden, N=256 gates). w_hh fragments pinned in
// registers; h kept in LDS as bf16; c/h state in thread registers.
// =====================================================================
__global__ __launch_bounds__(256) void k_lstm(
    const float* __restrict__ seq0, const float* __restrict__ seq1,
    const float* __restrict__ wih0, const float* __restrict__ bih0, const float* __restrict__ bhh0,
    const float* __restrict__ wih1, const float* __restrict__ bih1, const float* __restrict__ bhh1,
    const unsigned short* __restrict__ whhB,
    float* __restrict__ h0out, float* __restrict__ h1out) {
  __shared__ unsigned short gbuf[64][256];             // gate accumulator (bf16)
  __shared__ __align__(16) unsigned short hbuf[64*64]; // h state (bf16)
  __shared__ __align__(16) float sbuf[256];            // seq[t] (64x4)
  __shared__ float wihs[1024];                         // w_ih (256x4)
  __shared__ float bsums[256];                         // b_ih + b_hh

  const int br = blockIdx.x;
  const float* seq = br ? seq1 : seq0;
  const float* wih = br ? wih1 : wih0;
  const float* bih = br ? bih1 : bih0;
  const float* bhh = br ? bhh1 : bhh0;
  float* hout = br ? h1out : h0out;
  const int T = br ? BIN1 : BIN0;

  const int tid = threadIdx.x;
  const int lane = tid & 31, wave = tid >> 5;
  const int lanehi = lane >> 4, lanelo = lane & 15;
  const int mtile = wave >> 1, nt0 = (wave & 1) * 8;

  for (int i = tid; i < 4096; i += 256) hbuf[i] = 0;
  for (int i = tid; i < 1024; i += 256) wihs[i] = wih[i];
  bsums[tid] = bih[tid] + bhh[tid];

  BFrag bfr[8][2];                                     // w_hh^T fragments, in registers
  const u32x4* wq = (const u32x4*)whhB;
  #pragma unroll
  for (int nt = 0; nt < 8; ++nt)
    #pragma unroll
    for (int kc = 0; kc < 2; ++kc) {
      int base = (((br*2 + kc)*16 + (nt0 + nt))*32 + lane)*2;
      bfr[nt][kc].q[0] = wq[base];
      bfr[nt][kc].q[1] = wq[base + 1];
    }

  float cst[16], hst[16];
  #pragma unroll
  for (int q = 0; q < 16; ++q) { cst[q] = 0.0f; hst[q] = 0.0f; }
  const int sn = tid >> 2, shb = (tid & 3) * 16;       // this thread's (batch, hidden base)
  __syncthreads();

  for (int t = 0; t < T; ++t) {
    if (tid < 64) async_copy_b128(&sbuf[tid*4], &seq[t*256 + tid*4]);
    async_wait0();
    __syncthreads();
    // xg = b_ih + b_hh + x_t @ w_ih^T   (K=4, VALU into gbuf)
    for (int idx = tid; idx < 64*256; idx += 256) {
      int nn = idx >> 8, g = idx & 255;
      float v = bsums[g];
      #pragma unroll
      for (int j = 0; j < 4; ++j) v += sbuf[nn*4 + j] * wihs[g*4 + j];
      gbuf[nn][g] = f2bf(v);
    }
    __syncthreads();

    // A = h (64x64 bf16): two K-chunks, 2x ds_load_b128 each
    const int mrowA = mtile*16 + lanelo;
    const u32x4* hrow = (const u32x4*)(hbuf + mrowA*64);
    BFrag a0, a1;
    a0.q[0] = hrow[lanehi];     a0.q[1] = hrow[2 + lanehi];
    a1.q[0] = hrow[4 + lanehi]; a1.q[1] = hrow[6 + lanehi];

    #pragma unroll
    for (int nt = 0; nt < 8; ++nt) {
      const int gcol = (nt0 + nt)*16 + lanelo;
      v8f c;
      #pragma unroll
      for (int r = 0; r < 8; ++r) c[r] = bf2f(gbuf[mtile*16 + r + 8*lanehi][gcol]);
      c = wmma_bf16(a0.v, bfr[nt][0].v, c);
      c = wmma_bf16(a1.v, bfr[nt][1].v, c);
      #pragma unroll
      for (int r = 0; r < 8; ++r) gbuf[mtile*16 + r + 8*lanehi][gcol] = f2bf(c[r]);
    }
    __syncthreads();

    // elementwise LSTM cell update (PyTorch gate order i,f,g,o)
    #pragma unroll
    for (int q = 0; q < 16; ++q) {
      int h = shb + q;
      float ig = sigmf(bf2f(gbuf[sn][h]));
      float fg = sigmf(bf2f(gbuf[sn][64 + h]));
      float gg = tanhf(bf2f(gbuf[sn][128 + h]));
      float og = sigmf(bf2f(gbuf[sn][192 + h]));
      float cc = fg*cst[q] + ig*gg;
      cst[q] = cc;
      float hh = og * tanhf(cc);
      hst[q] = hh;
      hbuf[sn*64 + h] = f2bf(hh);
    }
    __syncthreads();
  }
  #pragma unroll
  for (int q = 0; q < 16; ++q) hout[sn*64 + shb + q] = hst[q];
}

// =====================================================================
// K_final: per-branch Linear(64->1), concat, Linear(2->1), sigmoid
// =====================================================================
__global__ void k_final(
    const float* __restrict__ h0, const float* __restrict__ h1,
    const float* __restrict__ wl0, const float* __restrict__ bl0,
    const float* __restrict__ wl1, const float* __restrict__ bl1,
    const float* __restrict__ wrul, const float* __restrict__ brul,
    float* __restrict__ out) {
  int n = threadIdx.x;
  if (n >= BATCH) return;
  float y0 = bl0[0], y1 = bl1[0];
  for (int k = 0; k < 64; ++k) {
    y0 += h0[n*64 + k] * wl0[k];
    y1 += h1[n*64 + k] * wl1[k];
  }
  float z = brul[0] + wrul[0]*y0 + wrul[1]*y1;
  out[n] = 1.0f / (1.0f + __expf(-z));
}

// =====================================================================
extern "C" void kernel_launch(void* const* d_in, const int* in_sizes, int n_in,
                              void* d_out, int out_size, void* d_ws, size_t ws_size,
                              hipStream_t stream) {
  (void)in_sizes; (void)n_in; (void)out_size; (void)ws_size;
  const float* X     = (const float*)d_in[0];
  const float* w_dw  = (const float*)d_in[1];
  const float* b_dw  = (const float*)d_in[2];
  const float* w_pw  = (const float*)d_in[3];
  const float* b_pw  = (const float*)d_in[4];
  const float* w_c2  = (const float*)d_in[5];
  const float* b_c2  = (const float*)d_in[6];
  const float* w_sc0 = (const float*)d_in[7];
  const float* b_sc0 = (const float*)d_in[8];
  const float* w_ih0 = (const float*)d_in[9];
  const float* b_ih0 = (const float*)d_in[10];
  const float* w_hh0 = (const float*)d_in[11];
  const float* b_hh0 = (const float*)d_in[12];
  const float* w_l0  = (const float*)d_in[13];
  const float* b_l0  = (const float*)d_in[14];
  const float* w_sc1 = (const float*)d_in[15];
  const float* b_sc1 = (const float*)d_in[16];
  const float* w_ih1 = (const float*)d_in[17];
  const float* b_ih1 = (const float*)d_in[18];
  const float* w_hh1 = (const float*)d_in[19];
  const float* b_hh1 = (const float*)d_in[20];
  const float* w_l1  = (const float*)d_in[21];
  const float* b_l1  = (const float*)d_in[22];
  const float* w_rul = (const float*)d_in[23];
  const float* b_rul = (const float*)d_in[24];

  char* ws = (char*)d_ws;
  float*          beff   = (float*)(ws + O_BEFF);
  unsigned short* w1B    = (unsigned short*)(ws + O_W1B);
  unsigned short* w2B    = (unsigned short*)(ws + O_W2B);
  unsigned short* whhB   = (unsigned short*)(ws + O_WHH);
  unsigned short* Xbf    = (unsigned short*)(ws + O_XBF);
  unsigned short* pbf    = (unsigned short*)(ws + O_PBF);
  float*          feat   = (float*)(ws + O_FEAT);
  float*          xp0    = (float*)(ws + O_XP0);
  float*          xp1    = (float*)(ws + O_XP1);
  float*          seq0   = (float*)(ws + O_SEQ0);
  float*          seq1   = (float*)(ws + O_SEQ1);
  float*          hb0    = (float*)(ws + O_H0);
  float*          hb1    = (float*)(ws + O_H1);

  // X -> bf16 (once per launch)
  k_xbf<<<(BATCH*16*LIN/2 + 255)/256, 256, 0, stream>>>(X, (unsigned int*)Xbf);
  // pack weights into WMMA fragment layouts (32 + 16384 + 32768 + 32768 items)
  k_prep<<<321, 256, 0, stream>>>(w_dw, b_dw, w_pw, b_pw, w_c2, w_hh0, w_hh1,
                                  beff, w1B, w2B, whhB);
  // fused dw+pw conv + relu + maxpool : 17 pool tiles x 64 batches
  k_conv1<<<dim3(17, BATCH), 256, 0, stream>>>(Xbf, w1B, beff, pbf);
  // conv2 + relu : 7 position tiles x 64 batches
  k_conv2<<<dim3(7, BATCH), 256, 0, stream>>>(pbf, w2B, b_c2, feat);
  // adaptive maxpool (both branches)
  k_apool<<<6400, 256, 0, stream>>>(feat, xp0, xp1);
  // branch convs 64->4 k=3 p=1 (both branches)
  k_conv3<<<400, 256, 0, stream>>>(xp0, xp1, w_sc0, b_sc0, w_sc1, b_sc1, seq0, seq1);
  // dual-branch LSTM, one workgroup each
  k_lstm<<<2, 256, 0, stream>>>(seq0, seq1,
                                w_ih0, b_ih0, b_hh0, w_ih1, b_ih1, b_hh1,
                                whhB, hb0, hb1);
  // heads + sigmoid
  k_final<<<1, 64, 0, stream>>>(hb0, hb1, w_l0, b_l0, w_l1, b_l1, w_rul, b_rul,
                                (float*)d_out);
}
